// MultiHeadAttention_85298050498565
// MI455X (gfx1250) — compile-verified, matrix operand order checked by reading, not run on hardware
//
#include <hip/hip_runtime.h>
#include <hip/hip_bf16.h>
#include <math.h>
#include <stdint.h>

// ---------------- problem constants ----------------
#define B_      2
#define S_      2048
#define E_      2048
#define H_      16
#define KVH_    4
#define D_      128
#define GROUPS_ (H_ / KVH_)
#define WINDOW_ 1024
#define CAP_    50.0f
#define RMS_EPS_ 1e-6f
#define LN_EPS_  1e-5f
#define M_      (B_ * S_)      // 4096 rows
#define KVE_    (KVH_ * D_)    // 512

typedef __bf16 bf16;
typedef __attribute__((ext_vector_type(8)))  bf16  bf16x8;
typedef __attribute__((ext_vector_type(16))) bf16  bf16x16;
typedef __attribute__((ext_vector_type(8)))  float f32x8;
typedef __attribute__((ext_vector_type(4)))  unsigned int u32x4;
typedef __attribute__((ext_vector_type(8)))  int   i32x8;
typedef __attribute__((ext_vector_type(4)))  int   i32x4;

// ---------------- gfx1250 feature detection (device pass only) --------------
#if defined(__gfx1250__)
# if __has_builtin(__builtin_amdgcn_tensor_load_to_lds)
#  define USE_TDM 1
# else
#  define USE_TDM 0
# endif
# if __has_builtin(__builtin_amdgcn_global_load_async_to_lds_b128)
#  define USE_ASYNC 1
# else
#  define USE_ASYNC 0
# endif
#else
# define USE_TDM 0
# define USE_ASYNC 0
#endif

#if USE_TDM
# if __has_include(<hip/amd_detail/amd_gfx1250_TDM.h>)
#  define TDM_ARGS6 1   // amdgpu-toolchain / therock headers -> 6-arg builtin
# else
#  define TDM_ARGS6 0   // ROCm 7.2 -> 5-arg builtin
# endif
#endif

__device__ __forceinline__ void wait_async0() {
#if defined(__gfx1250__)
# if __has_builtin(__builtin_amdgcn_s_wait_asynccnt)
  __builtin_amdgcn_s_wait_asynccnt(0);
# else
  asm volatile("s_wait_asynccnt 0x0" ::: "memory");
# endif
#endif
}

__device__ __forceinline__ void wait_tensor0() {
#if defined(__gfx1250__)
# if __has_builtin(__builtin_amdgcn_s_wait_tensorcnt)
  __builtin_amdgcn_s_wait_tensorcnt(0);
# else
  asm volatile("s_wait_tensorcnt 0x0" ::: "memory");
# endif
#endif
}

#if USE_ASYNC
// One 16B chunk: global -> LDS on the ASYNCcnt path (no VGPR staging).
// Builtin signature (from hipcc diagnostic): first param is
// 'int __vector(4) addrspace(1)*' (global src); LDS dst is the AS(3) analogue.
typedef int v4i_vs __attribute__((vector_size(16)));
__device__ __forceinline__ void async_copy_b128(void* lds, const void* g) {
  typedef __attribute__((address_space(1))) v4i_vs gvec_t;
  typedef __attribute__((address_space(3))) v4i_vs lvec_t;
  __builtin_amdgcn_global_load_async_to_lds_b128(
      (gvec_t*)(uintptr_t)g, (lvec_t*)(unsigned)(uintptr_t)lds, 0, 0);
}
#endif

#if USE_TDM
// Issue a TDM 2D tile load (bf16 elements) into LDS with optional row padding.
// D# layout per CDNA5 ISA ch.8: group0 = {ctl, lds_addr, global_addr, type},
// group1 = {flags/pad, tensor dims, tile dims, dim0 stride}. pad codes:
// pad_interval p -> pad every 2^(p+1) DWORDs, pad_amount a -> (a+1) DWORDs.
__device__ __forceinline__ void tdm_load_2d(
    unsigned lds_addr, const void* gaddr, unsigned tensor_d0,
    unsigned tensor_d1, unsigned long long stride0_elems, unsigned tile_d0,
    unsigned tile_d1, unsigned pad_interval, unsigned pad_amount) {
  unsigned long long ga = (unsigned long long)(uintptr_t)gaddr;
  u32x4 g0;
  g0[0] = 1u;                                   // count=1, user descriptor
  g0[1] = lds_addr;                             // LDS byte address
  g0[2] = (unsigned)(ga & 0xFFFFFFFFu);         // global_addr[31:0]
  g0[3] = (unsigned)((ga >> 32) & 0x01FFFFFFu)  // global_addr[56:32]
          | (2u << 30);                         // type=2 ("image")
  i32x8 g1;
  unsigned f = (1u << 16)                       // data_size = 2 bytes
             | (1u << 20)                       // pad_enable
             | ((pad_interval & 7u) << 22)
             | ((pad_amount & 127u) << 25);
  g1[0] = (int)f;                               // mask=0 (not in cluster)
  g1[1] = (int)((tensor_d0 & 0xFFFFu) << 16);   // tensor_dim0[15:0] @ 63:48
  g1[2] = (int)(((tensor_d0 >> 16) & 0xFFFFu) | ((tensor_d1 & 0xFFFFu) << 16));
  g1[3] = (int)(((tensor_d1 >> 16) & 0xFFFFu) | ((tile_d0 & 0xFFFFu) << 16));
  g1[4] = (int)(tile_d1 & 0xFFFFu);             // tile_dim1; tile_dim2=0
  g1[5] = (int)(unsigned)(stride0_elems & 0xFFFFFFFFull);
  g1[6] = (int)(unsigned)((stride0_elems >> 32) & 0xFFFFull);
  g1[7] = 0;
  i32x4 z4 = {0, 0, 0, 0};
# if TDM_ARGS6
  i32x8 z8 = {0, 0, 0, 0, 0, 0, 0, 0};
  __builtin_amdgcn_tensor_load_to_lds(g0, g1, z4, z4, z8, 0);
# else
  __builtin_amdgcn_tensor_load_to_lds(g0, g1, z4, z4, 0);
# endif
}
#endif

// Load a 16-element bf16 A/B fragment: 8 contiguous bf16 at p, 8 at p+16.
// Matches CDNA5 WMMA 16-bit operand layout (lane half selects K sub-block).
__device__ __forceinline__ bf16x16 ld_frag(const bf16* p) {
  bf16x8 lo = *(const bf16x8*)p;
  bf16x8 hi = *(const bf16x8*)(p + 16);
  return __builtin_shufflevector(lo, hi, 0, 1, 2, 3, 4, 5, 6, 7,
                                 8, 9, 10, 11, 12, 13, 14, 15);
}

// ---------------- f32 -> bf16 downcast ----------------
__global__ __launch_bounds__(256) void cast_bf16_kernel(
    const float* __restrict__ src, bf16* __restrict__ dst, int n) {
  int i = blockIdx.x * blockDim.x + threadIdx.x;
  int stride = gridDim.x * blockDim.x;
  for (; i < n; i += stride) dst[i] = (bf16)src[i];
}

// ---------------- bf16 GEMM: C[M,N] = A[M,K] * B[N,K]^T (+R) ----------------
// 128x128x32 LDS tiling, 8 waves, each wave does a 32x64 sub-tile = 2x4 WMMAs.
// Tile fills go through the ASYNCcnt path when available.
__global__ __launch_bounds__(256) void gemm_bf16_kernel(
    const bf16* __restrict__ A, const bf16* __restrict__ Bm,
    const float* __restrict__ R, float* __restrict__ C,
    int Mdim, int Ndim, int Kdim) {
  __shared__ __align__(16) bf16 As[128][40];  // 32 bf16 + 8 pad (80B rows)
  __shared__ __align__(16) bf16 Bs[128][40];

  const int t    = threadIdx.x;
  const int lane = t & 31;
  const int w    = t >> 5;
  const int half = lane >> 4;
  const int l16  = lane & 15;
  const int m0   = blockIdx.y * 128;
  const int n0   = blockIdx.x * 128;
  const int wm   = w & 3;   // 4 waves along M
  const int wn   = w >> 2;  // 2 waves along N

  f32x8 acc[2][4];
  const f32x8 zero8 = {0.f, 0.f, 0.f, 0.f, 0.f, 0.f, 0.f, 0.f};
  for (int i = 0; i < 2; ++i)
    for (int j = 0; j < 4; ++j) acc[i][j] = zero8;

  for (int k0 = 0; k0 < Kdim; k0 += 32) {
    // cooperative tile fill: 128 rows x 32 bf16 per matrix, 16B chunks
    for (int p = 0; p < 2; ++p) {
      int c   = t + p * 256;  // 512 chunks of 16B
      int row = c >> 2;
      int seg = c & 3;
      const bf16* ga = &A[(size_t)(m0 + row) * Kdim + k0 + seg * 8];
      const bf16* gb = &Bm[(size_t)(n0 + row) * Kdim + k0 + seg * 8];
#if USE_ASYNC
      async_copy_b128(&As[row][seg * 8], ga);
      async_copy_b128(&Bs[row][seg * 8], gb);
#else
      *(bf16x8*)&As[row][seg * 8] = *(const bf16x8*)ga;
      *(bf16x8*)&Bs[row][seg * 8] = *(const bf16x8*)gb;
#endif
    }
    if (k0 + 32 < Kdim) {  // global_prefetch_b8 of next K tile
      __builtin_prefetch(&A[(size_t)(m0 + (t >> 1)) * Kdim + k0 + 32], 0, 1);
      __builtin_prefetch(&Bm[(size_t)(n0 + (t >> 1)) * Kdim + k0 + 32], 0, 1);
    }
#if USE_ASYNC
    wait_async0();
#endif
    __syncthreads();

    bf16x16 af[2], bfr[4];
    for (int i = 0; i < 2; ++i)
      af[i] = ld_frag(&As[wm * 32 + i * 16 + l16][half * 8]);
    for (int j = 0; j < 4; ++j)
      bfr[j] = ld_frag(&Bs[wn * 64 + j * 16 + l16][half * 8]);

    for (int i = 0; i < 2; ++i)
      for (int j = 0; j < 4; ++j)
        acc[i][j] = __builtin_amdgcn_wmma_f32_16x16x32_bf16(
            false, af[i], false, bfr[j], (short)0, acc[i][j], false, false);
    __syncthreads();
  }

  // epilogue: C layout row = r + 8*half, col = l16 (coalesced 16-lane runs)
  for (int i = 0; i < 2; ++i)
    for (int j = 0; j < 4; ++j)
      for (int r = 0; r < 8; ++r) {
        int row = m0 + wm * 32 + i * 16 + r + half * 8;
        int col = n0 + wn * 64 + j * 16 + l16;
        size_t idx = (size_t)row * Ndim + col;
        float v = acc[i][j][r];
        if (R) v += R[idx];
        C[idx] = v;
      }
}

// ---------------- per-head RMSNorm + RoPE, [B,S,nh,D] -> bf16 [B,nh,S,D] ----
__global__ __launch_bounds__(128) void rmsrope_kernel(
    const float* __restrict__ in, const float* __restrict__ wt,
    bf16* __restrict__ out, int nh) {
  const int d   = threadIdx.x;  // 0..127
  const int idx = blockIdx.x;   // (b*S + s)*nh + h
  const int h   = idx % nh;
  const int s   = (idx / nh) % S_;
  const int b   = idx / (nh * S_);

  __shared__ float nt[D_];
  __shared__ float red[4];

  float v  = in[(size_t)idx * D_ + d];
  float ss = v * v;
  for (int m = 1; m < 32; m <<= 1) ss += __shfl_xor(ss, m, 32);
  if ((threadIdx.x & 31) == 0) red[threadIdx.x >> 5] = ss;
  __syncthreads();
  float tot = red[0] + red[1] + red[2] + red[3];
  float rms = rsqrtf(tot * (1.0f / D_) + RMS_EPS_);
  float nv  = v * rms * wt[d];
  nt[d] = nv;
  __syncthreads();

  int   j   = d & 63;
  float ang = (float)s * powf(10000.0f, -(float)j * (1.0f / 64.0f));
  float c   = cosf(ang);
  float sn  = sinf(ang);
  float rot = (d < 64) ? -nt[d + 64] : nt[d - 64];
  float o   = nv * c + rot * sn;
  out[(((size_t)b * nh + h) * S_ + s) * D_ + d] = (bf16)o;
}

// ---------------- V transpose: f32 [B,S,KVH,D] -> bf16 [B,KVH,D,S] ----------
__global__ __launch_bounds__(256) void v_transpose_kernel(
    const float* __restrict__ v, bf16* __restrict__ vt) {
  int i = blockIdx.x * blockDim.x + threadIdx.x;
  int n = B_ * S_ * KVH_ * D_;
  int stride = gridDim.x * blockDim.x;
  for (; i < n; i += stride) {
    int d = i % D_;
    int g = (i / D_) % KVH_;
    int s = (i / (D_ * KVH_)) % S_;
    int b = i / (D_ * KVH_ * S_);
    vt[(((size_t)b * KVH_ + g) * D_ + d) * S_ + s] = (bf16)v[i];
  }
}

// ---------------- flash attention: sliding window + tanh cap (GQA) ----------
// Block = 256 threads = 8 waves; block handles (b, h, 128 query rows);
// wave owns 16 query rows. K/V tiles of 32 keys staged in LDS; on gfx1250
// the tiles are filled by the Tensor Data Mover (one wave issues the DMA,
// TENSORcnt tracks completion, the TDM pad feature reproduces the padded
// LDS strides the WMMA fragment loads expect).
__global__ __launch_bounds__(256) void attn_kernel(
    const bf16* __restrict__ Q,   // [B,H,S,D]
    const bf16* __restrict__ K,   // [B,KVH,S,D]
    const bf16* __restrict__ Vt,  // [B,KVH,D,S]
    bf16* __restrict__ O) {       // [B,S,E]
  __shared__ __align__(16) bf16 Ks[32][136];    // keys x d  (272B rows)
  __shared__ __align__(16) bf16 Vs[128][40];    // d x keys  (80B rows)
  __shared__ __align__(16) bf16 Ps[8][16][40];  // per-wave P staging

  const int t    = threadIdx.x;
  const int lane = t & 31;
  const int w    = t >> 5;
  const int half = lane >> 4;
  const int l16  = lane & 15;
  const int q0   = blockIdx.x * 128;
  const int h    = blockIdx.y;
  const int b    = blockIdx.z;
  const int g    = h / GROUPS_;
  const int qb   = q0 + w * 16;

  const bf16* Kbase = K + ((size_t)b * KVH_ + g) * (size_t)S_ * D_;
  const bf16* Vbase = Vt + ((size_t)b * KVH_ + g) * (size_t)D_ * S_;

  // Q fragments (held in registers for the whole key loop): 4 K-blocks of 32
  bf16x16 qf[4];
  {
    const bf16* qp = Q + (((size_t)b * H_ + h) * S_ + qb + l16) * D_;
    for (int dk = 0; dk < 4; ++dk) qf[dk] = ld_frag(qp + dk * 32 + half * 8);
  }

  const f32x8 zero8 = {0.f, 0.f, 0.f, 0.f, 0.f, 0.f, 0.f, 0.f};
  f32x8 o[8];
  for (int j = 0; j < 8; ++j) o[j] = zero8;
  float mrow[8], lrow[8];
  for (int r = 0; r < 8; ++r) { mrow[r] = -3.0e38f; lrow[r] = 0.0f; }

  int kbeg = q0 - (WINDOW_ - 1);
  if (kbeg < 0) kbeg = 0;
  kbeg &= ~31;
  const int kend    = q0 + 127;
  const float scale = 0.088388347648318447f;  // 1/sqrt(128)
  const float l2e   = 1.4426950408889634f;

  for (int kc = kbeg; kc <= kend; kc += 32) {
#if USE_TDM
    if (w == 0) {
      // K tile: 32 rows x 128 bf16 (256B) + 16B pad -> Ks[32][136]
      tdm_load_2d((unsigned)(uintptr_t)&Ks[0][0], Kbase + (size_t)kc * D_,
                  /*tensor_d0=*/D_, /*tensor_d1=*/(unsigned)(S_ - kc),
                  /*stride0=*/D_, /*tile_d0=*/D_, /*tile_d1=*/32,
                  /*pad_interval=*/5, /*pad_amount=*/3);
      // V tile: 128 rows x 32 bf16 (64B) + 16B pad -> Vs[128][40]
      tdm_load_2d((unsigned)(uintptr_t)&Vs[0][0], Vbase + kc,
                  /*tensor_d0=*/(unsigned)(S_ - kc), /*tensor_d1=*/D_,
                  /*stride0=*/S_, /*tile_d0=*/32, /*tile_d1=*/D_,
                  /*pad_interval=*/3, /*pad_amount=*/3);
      wait_tensor0();
    }
#else
    // cooperative K/V tile fill (16B chunks)
    for (int p = 0; p < 2; ++p) {
      int c = t + p * 256;
      {
        int row = c >> 4, seg = c & 15;  // 32 rows x 16 chunks
        *(bf16x8*)&Ks[row][seg * 8] =
            *(const bf16x8*)&Kbase[(size_t)(kc + row) * D_ + seg * 8];
      }
      {
        int dr = c >> 2, seg = c & 3;    // 128 rows x 4 chunks
        *(bf16x8*)&Vs[dr][seg * 8] =
            *(const bf16x8*)&Vbase[(size_t)dr * S_ + kc + seg * 8];
      }
    }
#endif
    __syncthreads();

    const bool active = !((kc + 31 < qb - (WINDOW_ - 1)) || (kc > qb + 15));
    if (active) {
      // ---- scores: S = Q * K^T over D=128 (4 chained WMMAs per key tile)
      f32x8 sacc[2] = {zero8, zero8};
      for (int tt = 0; tt < 2; ++tt)
        for (int dk = 0; dk < 4; ++dk) {
          bf16x16 kf = ld_frag(&Ks[tt * 16 + l16][dk * 32 + half * 8]);
          sacc[tt] = __builtin_amdgcn_wmma_f32_16x16x32_bf16(
              false, qf[dk], false, kf, (short)0, sacc[tt], false, false);
        }

      // ---- online softmax with tanh soft-cap + sliding-window mask
      float pval[2][8];
      for (int r = 0; r < 8; ++r) {
        const int qi = qb + r + half * 8;
        float x[2];
        float mx = -3.0e38f;
        for (int tt = 0; tt < 2; ++tt) {
          int   j  = kc + tt * 16 + l16;
          float sr = sacc[tt][r] * scale;
          sr = CAP_ * tanhf(sr * (1.0f / CAP_));
          bool ok = (j <= qi) && (qi - j < WINDOW_);
          x[tt] = ok ? sr : -3.0e38f;
          mx = fmaxf(mx, x[tt]);
        }
        for (int m = 1; m < 16; m <<= 1) mx = fmaxf(mx, __shfl_xor(mx, m, 32));
        float mN    = fmaxf(mrow[r], mx);
        float alpha = exp2f((mrow[r] - mN) * l2e);
        float psum  = 0.0f;
        for (int tt = 0; tt < 2; ++tt) {
          float p = (x[tt] <= -1.0e37f) ? 0.0f : exp2f((x[tt] - mN) * l2e);
          pval[tt][r] = p;
          psum += p;
        }
        for (int m = 1; m < 16; m <<= 1) psum += __shfl_xor(psum, m, 32);
        lrow[r] = lrow[r] * alpha + psum;
        mrow[r] = mN;
        for (int jj = 0; jj < 8; ++jj) o[jj][r] *= alpha;
      }

      // ---- transpose P from C-layout to A-layout via per-wave LDS staging
      for (int r = 0; r < 8; ++r) {
        int prow = r + half * 8;
        Ps[w][prow][l16]      = (bf16)pval[0][r];
        Ps[w][prow][16 + l16] = (bf16)pval[1][r];
      }
      bf16x16 pf = ld_frag(&Ps[w][l16][half * 8]);

      // ---- O += P * V (8 output d-tiles, contraction over 32 keys)
      for (int jj = 0; jj < 8; ++jj) {
        bf16x16 vf = ld_frag(&Vs[jj * 16 + l16][half * 8]);
        o[jj] = __builtin_amdgcn_wmma_f32_16x16x32_bf16(
            false, pf, false, vf, (short)0, o[jj], false, false);
      }
    }
    __syncthreads();
  }

  // normalize + store bf16 attention output [B,S,E]
  for (int jj = 0; jj < 8; ++jj)
    for (int r = 0; r < 8; ++r) {
      int   srow = qb + r + half * 8;
      float val  = o[jj][r] / lrow[r];
      O[((size_t)(b * S_ + srow)) * E_ + h * D_ + jj * 16 + l16] = (bf16)val;
    }
}

// ---------------- final LayerNorm over E per row ----------------
__global__ __launch_bounds__(256) void layernorm_kernel(
    const float* __restrict__ Y, const float* __restrict__ gamma,
    const float* __restrict__ beta, float* __restrict__ out) {
  const int row  = blockIdx.x;
  const float* y = Y + (size_t)row * E_;
  float s1 = 0.f, s2 = 0.f;
  for (int i = threadIdx.x; i < E_; i += 256) {
    float v = y[i];
    s1 += v;
    s2 += v * v;
  }
  for (int m = 1; m < 32; m <<= 1) {
    s1 += __shfl_xor(s1, m, 32);
    s2 += __shfl_xor(s2, m, 32);
  }
  __shared__ float a1[8], a2[8];
  if ((threadIdx.x & 31) == 0) {
    a1[threadIdx.x >> 5] = s1;
    a2[threadIdx.x >> 5] = s2;
  }
  __syncthreads();
  float t1 = 0.f, t2 = 0.f;
  for (int i = 0; i < 8; ++i) { t1 += a1[i]; t2 += a2[i]; }
  float mu  = t1 * (1.0f / E_);
  float var = t2 * (1.0f / E_) - mu * mu;
  float inv = rsqrtf(var + LN_EPS_);
  for (int i = threadIdx.x; i < E_; i += 256)
    out[(size_t)row * E_ + i] = (y[i] - mu) * inv * gamma[i] + beta[i];
}

// ---------------- host-side launch ----------------
extern "C" void kernel_launch(void* const* d_in, const int* in_sizes, int n_in,
                              void* d_out, int out_size, void* d_ws,
                              size_t ws_size, hipStream_t stream) {
  (void)in_sizes; (void)n_in; (void)out_size; (void)ws_size;
  const float* x   = (const float*)d_in[0];  // [B,S,E]
  const float* Wq  = (const float*)d_in[1];  // [E,E]
  const float* Wk  = (const float*)d_in[2];  // [KVE,E]
  const float* Wv  = (const float*)d_in[3];  // [KVE,E]
  const float* Wo  = (const float*)d_in[4];  // [E,E]
  const float* qnw = (const float*)d_in[5];  // [D]
  const float* knw = (const float*)d_in[6];  // [D]
  const float* lng = (const float*)d_in[7];  // [E]
  const float* lnb = (const float*)d_in[8];  // [E]
  float*       out = (float*)d_out;

  // workspace carve-out (256B aligned)
  char* ws   = (char*)d_ws;
  size_t off = 0;
  auto carve = [&](size_t bytes) -> char* {
    char* p = ws + off;
    off = (off + bytes + 255) & ~(size_t)255;
    return p;
  };
  bf16*  xb    = (bf16*)carve((size_t)M_ * E_ * 2);
  bf16*  Wqb   = (bf16*)carve((size_t)E_ * E_ * 2);
  bf16*  Wkb   = (bf16*)carve((size_t)KVE_ * E_ * 2);
  bf16*  Wvb   = (bf16*)carve((size_t)KVE_ * E_ * 2);
  bf16*  Wob   = (bf16*)carve((size_t)E_ * E_ * 2);
  float* qf    = (float*)carve((size_t)M_ * E_ * 4);
  float* kf    = (float*)carve((size_t)M_ * KVE_ * 4);
  float* vf    = (float*)carve((size_t)M_ * KVE_ * 4);
  bf16*  qbf   = (bf16*)carve((size_t)M_ * E_ * 2);    // [B,H,S,D]
  bf16*  kbf   = (bf16*)carve((size_t)M_ * KVE_ * 2);  // [B,KVH,S,D]
  bf16*  vtb   = (bf16*)carve((size_t)M_ * KVE_ * 2);  // [B,KVH,D,S]
  bf16*  attnb = (bf16*)carve((size_t)M_ * E_ * 2);    // [B,S,E]
  float* ypre  = (float*)carve((size_t)M_ * E_ * 4);

  // 1) downcast inputs to bf16
  cast_bf16_kernel<<<2048, 256, 0, stream>>>(x, xb, M_ * E_);
  cast_bf16_kernel<<<2048, 256, 0, stream>>>(Wq, Wqb, E_ * E_);
  cast_bf16_kernel<<<1024, 256, 0, stream>>>(Wk, Wkb, KVE_ * E_);
  cast_bf16_kernel<<<1024, 256, 0, stream>>>(Wv, Wvb, KVE_ * E_);
  cast_bf16_kernel<<<2048, 256, 0, stream>>>(Wo, Wob, E_ * E_);

  // 2) QKV projections (WMMA bf16, f32 accumulate)
  gemm_bf16_kernel<<<dim3(E_ / 128, M_ / 128), 256, 0, stream>>>(
      xb, Wqb, nullptr, qf, M_, E_, E_);
  gemm_bf16_kernel<<<dim3(KVE_ / 128, M_ / 128), 256, 0, stream>>>(
      xb, Wkb, nullptr, kf, M_, KVE_, E_);
  gemm_bf16_kernel<<<dim3(KVE_ / 128, M_ / 128), 256, 0, stream>>>(
      xb, Wvb, nullptr, vf, M_, KVE_, E_);

  // 3) RMSNorm + RoPE (q,k) and V transpose, all to bf16 attention layouts
  rmsrope_kernel<<<B_ * S_ * H_, 128, 0, stream>>>(qf, qnw, qbf, H_);
  rmsrope_kernel<<<B_ * S_ * KVH_, 128, 0, stream>>>(kf, knw, kbf, KVH_);
  v_transpose_kernel<<<4096, 256, 0, stream>>>(vf, vtb);

  // 4) flash attention (sliding window 1024, tanh cap 50, GQA 4:1)
  attn_kernel<<<dim3(S_ / 128, H_, B_), 256, 0, stream>>>(qbf, kbf, vtb, attnb);

  // 5) output projection + residual, then LayerNorm
  gemm_bf16_kernel<<<dim3(E_ / 128, M_ / 128), 256, 0, stream>>>(
      attnb, Wob, x, ypre, M_, E_, E_);
  layernorm_kernel<<<M_, 256, 0, stream>>>(ypre, lng, lnb, out);
}
